// MultiStepGCNGRU_71159018160982
// MI455X (gfx1250) — compile-verified, matrix operand order checked by reading, not run on hardware
//
#include <hip/hip_runtime.h>
#include <hip/hip_bf16.h>

#define N_NODES 20000
#define N_EDGES 640000
#define IN_C 64
#define HID 256
#define OUT_C 64
#define HIST 12
#define HORIZON 6

typedef __attribute__((ext_vector_type(16))) __bf16 v16bf;
typedef __attribute__((ext_vector_type(8)))  float  v8f;

__device__ __forceinline__ unsigned short f2bf(float f) {
    unsigned int u = __float_as_uint(f);
    u += 0x7FFFu + ((u >> 16) & 1u);           // round-to-nearest-even
    return (unsigned short)(u >> 16);
}

// ---------------------------------------------------------------------------
// Generic utility kernels
// ---------------------------------------------------------------------------
__global__ void zero_f32(float* p, int n) {
    int i = blockIdx.x * blockDim.x + threadIdx.x;
    if (i < n) p[i] = 0.0f;
}

__global__ void degree_kernel(const int* __restrict__ dst, float* __restrict__ degcnt) {
    int e = blockIdx.x * blockDim.x + threadIdx.x;
    if (e < N_EDGES) atomicAdd(&degcnt[dst[e]], 1.0f);
}

// degcnt holds raw in-degree counts; produce deg_is = (deg+1)^-1/2 and deg_inv = 1/(deg+1)
__global__ void deg_finalize(float* __restrict__ deg_inv, float* __restrict__ deg_is) {
    int i = blockIdx.x * blockDim.x + threadIdx.x;
    if (i >= N_NODES) return;
    float d = deg_inv[i] + 1.0f;
    deg_is[i] = rsqrtf(d);
    deg_inv[i] = 1.0f / d;
}

__global__ void norm_kernel(const int* __restrict__ src, const int* __restrict__ dst,
                            const float* __restrict__ deg_is, float* __restrict__ norm) {
    int e = blockIdx.x * blockDim.x + threadIdx.x;
    if (e < N_EDGES) norm[e] = deg_is[src[e]] * deg_is[dst[e]];
}

// ---------------------------------------------------------------------------
// Weight packing: fp32 weight -> bf16 B-fragments for v_wmma_f32_16x16x32_bf16
// Logical B is [K,N]; trans==1 means source W is [N,K] (i.e. B = W^T).
// Fragment (kt,nt) = 32x16 block; lane L<16 -> col nt*16+L, K base kt*32+0;
// lane>=16 -> col nt*16+(L-16), K base kt*32+8. Element e: k = base + (e<8 ? e : e+8).
// Storage: out[(frag*32 + lane)*16 + e], frag = kt*(N/16)+nt  (32B per lane, contiguous)
// ---------------------------------------------------------------------------
__global__ void pack_b_kernel(const float* __restrict__ W, unsigned short* __restrict__ out,
                              int K, int N, int trans) {
    int tid = blockIdx.x * blockDim.x + threadIdx.x;
    int nt16 = N >> 4;
    int nfrag = (K >> 5) * nt16;
    int frag = tid >> 5;
    if (frag >= nfrag) return;
    int lane = tid & 31;
    int kt = frag / nt16;
    int nt = frag - kt * nt16;
    int n  = nt * 16 + (lane & 15);
    int kb = kt * 32 + ((lane >> 4) << 3);
    unsigned short* o = out + ((size_t)frag * 32 + lane) * 16;
#pragma unroll
    for (int e = 0; e < 16; ++e) {
        int k = kb + (e < 8 ? e : e + 8);
        float v = trans ? W[(size_t)n * K + k] : W[(size_t)k * N + n];
        o[e] = f2bf(v);
    }
}

// ---------------------------------------------------------------------------
// WMMA GEMM: C[M,N] = A[M,K](fp32) * Bpacked[K,N](bf16 fragments) (+bias[N])
// block: 256 threads = 8 waves; tile 256(M) x 64(N), K-step 32.
// Each wave owns 32 rows (2 A fragments) and reuses 4 B fragments -> 8 WMMA/iter.
// Double-buffered LDS A tile: one barrier per K-step; next tile's global loads
// overlap the current tile's WMMAs.
// ---------------------------------------------------------------------------
#define GTM 256
#define GTN 64
#define GTK 32
#define LDSTRIDE (GTK + 8)   // bf16 elems; 80B rows keep 16B alignment, skew banks

__global__ __launch_bounds__(256)
void gemm_bf16_wmma(const float* __restrict__ A, const unsigned short* __restrict__ Bp,
                    float* __restrict__ C, const float* __restrict__ bias,
                    int M, int K, int N) {
    __shared__ unsigned short ldsA[2][GTM * LDSTRIDE];
    const int tid  = threadIdx.x;
    const int wave = tid >> 5;
    const int lane = tid & 31;
    const int blockM = blockIdx.x * GTM;
    const int blockN = blockIdx.y * GTN;
    const int nt16 = N >> 4;

    v8f zero = {0.f, 0.f, 0.f, 0.f, 0.f, 0.f, 0.f, 0.f};
    v8f acc[2][4];
#pragma unroll
    for (int s = 0; s < 2; ++s)
#pragma unroll
        for (int t = 0; t < 4; ++t) acc[s][t] = zero;

    // cooperative A tile loader: 256 rows x 32 cols fp32 -> bf16 into buffer `buf`
    auto loadA = [&](int kt, int buf) {
#pragma unroll
        for (int i = 0; i < 8; ++i) {
            int f   = i * 256 + tid;       // float4 index within tile (0..2047)
            int row = f >> 3;              // 0..255
            int c4  = (f & 7) << 2;        // 0,4,...,28
            int grow = blockM + row;
            float4 v = make_float4(0.f, 0.f, 0.f, 0.f);
            if (grow < M)
                v = *(const float4*)(A + (size_t)grow * K + kt * GTK + c4);
            unsigned int* d = (unsigned int*)&ldsA[buf][row * LDSTRIDE + c4];
            d[0] = (unsigned int)f2bf(v.x) | ((unsigned int)f2bf(v.y) << 16);
            d[1] = (unsigned int)f2bf(v.z) | ((unsigned int)f2bf(v.w) << 16);
        }
    };

    const int kTiles = K >> 5;
    loadA(0, 0);
    __syncthreads();

    for (int kt = 0; kt < kTiles; ++kt) {
        const int cur = kt & 1;
        if (kt + 1 < kTiles) loadA(kt + 1, cur ^ 1);   // prefetch into other buffer

        // ---- 2 A fragments (rows wave*32 .. wave*32+31), per ISA 16-bit A layout
        const int m  = lane & 15;
        const int kb = (lane >> 4) << 3;               // 0 or 8
        const unsigned short* a0 = &ldsA[cur][(wave * 32 + m) * LDSTRIDE + kb];
        const unsigned short* a1 = a0 + 16 * LDSTRIDE;
        v16bf af0, af1;
        ((uint4*)&af0)[0] = *(const uint4*)(a0);
        ((uint4*)&af0)[1] = *(const uint4*)(a0 + 16);
        ((uint4*)&af1)[0] = *(const uint4*)(a1);
        ((uint4*)&af1)[1] = *(const uint4*)(a1 + 16);

        // ---- 4 B fragments from packed global (hot in L2) ----
        v16bf bf[4];
#pragma unroll
        for (int t = 0; t < 4; ++t) {
            int nt = (blockN >> 4) + t;
            const unsigned short* bp = Bp + ((size_t)(kt * nt16 + nt) * 32 + lane) * 16;
            ((uint4*)&bf[t])[0] = *(const uint4*)(bp);
            ((uint4*)&bf[t])[1] = *(const uint4*)(bp + 8);
        }

        // ---- 8 WMMAs: B fragments reused across both A fragments ----
#pragma unroll
        for (int t = 0; t < 4; ++t) {
            acc[0][t] = __builtin_amdgcn_wmma_f32_16x16x32_bf16(
                false, af0, false, bf[t], (short)0, acc[0][t], false, false);
            acc[1][t] = __builtin_amdgcn_wmma_f32_16x16x32_bf16(
                false, af1, false, bf[t], (short)0, acc[1][t], false, false);
        }
        __syncthreads();
    }

    // ---- epilogue: C/D layout -> global, optional bias ----
    const int col0 = blockN + (lane & 15);
#pragma unroll
    for (int s = 0; s < 2; ++s) {
        const int rowbase = blockM + wave * 32 + s * 16 + ((lane >> 4) << 3);
#pragma unroll
        for (int t = 0; t < 4; ++t) {
            int colg = col0 + t * 16;
            float bv = bias ? bias[colg] : 0.0f;
#pragma unroll
            for (int r = 0; r < 8; ++r) {
                int row = rowbase + r;
                if (row < M) C[(size_t)row * N + colg] = acc[s][t][r] + bv;
            }
        }
    }
}

// ---------------------------------------------------------------------------
// Edge aggregation: agg[dst] += xw[src] * norm  (one block per edge, 256 feats)
// ---------------------------------------------------------------------------
__global__ __launch_bounds__(256)
void edge_agg_kernel(const float* __restrict__ xw, const int* __restrict__ src,
                     const int* __restrict__ dst, const float* __restrict__ norm,
                     float* __restrict__ agg) {
    int e = blockIdx.x;
    int t = threadIdx.x;
    int s = src[e], d = dst[e];
    float w = norm[e];
    atomicAdd(&agg[(size_t)d * HID + t], xw[(size_t)s * HID + t] * w);
}

// x_out = relu(agg + xw*deg_inv + b)
__global__ void gcn_finish_kernel(const float* __restrict__ agg, const float* __restrict__ xw,
                                  const float* __restrict__ deg_inv, const float* __restrict__ b,
                                  float* __restrict__ out) {
    int i = blockIdx.x * blockDim.x + threadIdx.x;
    if (i >= N_NODES * HID) return;
    int node = i >> 8;
    int f = i & 255;
    float v = agg[i] + xw[i] * deg_inv[node] + b[f];
    out[i] = fmaxf(v, 0.0f);
}

// GRU gates: gi/gh hold [N,768] with bias already added in GEMM epilogue
__global__ void gru_gate_kernel(const float* __restrict__ gi, const float* __restrict__ gh,
                                const float* __restrict__ h, float* __restrict__ hout) {
    int i = blockIdx.x * blockDim.x + threadIdx.x;
    if (i >= N_NODES * HID) return;
    int node = i >> 8;
    int f = i & 255;
    size_t base = (size_t)node * (3 * HID) + f;
    float ir = gi[base], iz = gi[base + HID], in = gi[base + 2 * HID];
    float hr = gh[base], hz = gh[base + HID], hn = gh[base + 2 * HID];
    float r = 1.0f / (1.0f + __expf(-(ir + hr)));
    float z = 1.0f / (1.0f + __expf(-(iz + hz)));
    float nn = tanhf(in + r * hn);
    hout[i] = (1.0f - z) * nn + z * h[i];
}

// ---------------------------------------------------------------------------
// Host orchestration
// ---------------------------------------------------------------------------
extern "C" void kernel_launch(void* const* d_in, const int* in_sizes, int n_in,
                              void* d_out, int out_size, void* d_ws, size_t ws_size,
                              hipStream_t stream) {
    (void)in_sizes; (void)n_in; (void)out_size; (void)ws_size;
    const float* x_seq  = (const float*)d_in[0];
    const int*   src    = (const int*)d_in[1];
    const int*   dst    = (const int*)d_in[2];
    const float* gcn1_w = (const float*)d_in[3];
    const float* gcn1_b = (const float*)d_in[4];
    const float* gcn2_w = (const float*)d_in[5];
    const float* gcn2_b = (const float*)d_in[6];
    const float* proj_w = (const float*)d_in[7];
    const float* proj_b = (const float*)d_in[8];
    const float* wih[3] = {(const float*)d_in[9],  (const float*)d_in[13], (const float*)d_in[17]};
    const float* whh[3] = {(const float*)d_in[10], (const float*)d_in[14], (const float*)d_in[18]};
    const float* bih[3] = {(const float*)d_in[11], (const float*)d_in[15], (const float*)d_in[19]};
    const float* bhh[3] = {(const float*)d_in[12], (const float*)d_in[16], (const float*)d_in[20]};
    float* out = (float*)d_out;

    // ---- workspace carve-up ----
    char* ws = (char*)d_ws;
    size_t off = 0;
    auto alloc_f = [&](size_t n) { float* p = (float*)(ws + off); off += n * sizeof(float); return p; };
    auto alloc_u = [&](size_t n) { unsigned short* p = (unsigned short*)(ws + off);
                                   off += ((n * sizeof(unsigned short) + 15) & ~(size_t)15); return p; };
    float* deg_inv = alloc_f(N_NODES);
    float* deg_is  = alloc_f(N_NODES);
    float* norm    = alloc_f(N_EDGES);
    float* xw      = alloc_f((size_t)N_NODES * HID);
    float* agg     = alloc_f((size_t)N_NODES * HID);
    float* x1      = alloc_f((size_t)N_NODES * HID);
    float* x2      = alloc_f((size_t)N_NODES * HID);
    float* gi      = alloc_f((size_t)N_NODES * 3 * HID);
    float* gh      = alloc_f((size_t)N_NODES * 3 * HID);
    float* h[3]    = {alloc_f((size_t)N_NODES * HID), alloc_f((size_t)N_NODES * HID),
                      alloc_f((size_t)N_NODES * HID)};
    unsigned short* pw_gcn1 = alloc_u((size_t)IN_C * HID);
    unsigned short* pw_gcn2 = alloc_u((size_t)HID * HID);
    unsigned short* pw_wih[3], *pw_whh[3];
    for (int g = 0; g < 3; ++g) { pw_wih[g] = alloc_u((size_t)HID * 3 * HID);
                                  pw_whh[g] = alloc_u((size_t)HID * 3 * HID); }
    unsigned short* pw_proj = alloc_u((size_t)HID * OUT_C);

    const int TPB = 256;
    auto grid1 = [](int n) { return dim3((n + 255) / 256); };

    // ---- one-time per-launch prep (deterministic) ----
    zero_f32<<<grid1(N_NODES), TPB, 0, stream>>>(deg_inv, N_NODES);   // reuse as count
    degree_kernel<<<grid1(N_EDGES), TPB, 0, stream>>>(dst, deg_inv);
    deg_finalize<<<grid1(N_NODES), TPB, 0, stream>>>(deg_inv, deg_is);
    norm_kernel<<<grid1(N_EDGES), TPB, 0, stream>>>(src, dst, deg_is, norm);

    auto pack = [&](const float* W, unsigned short* o, int K, int N, int trans) {
        int threads = (K / 32) * (N / 16) * 32;
        pack_b_kernel<<<grid1(threads), TPB, 0, stream>>>(W, o, K, N, trans);
    };
    pack(gcn1_w, pw_gcn1, IN_C, HID, 0);
    pack(gcn2_w, pw_gcn2, HID, HID, 0);
    for (int g = 0; g < 3; ++g) {
        pack(wih[g], pw_wih[g], HID, 3 * HID, 1);   // wih is [3H, H] -> B = wih^T
        pack(whh[g], pw_whh[g], HID, 3 * HID, 1);
    }
    pack(proj_w, pw_proj, HID, OUT_C, 0);

    for (int g = 0; g < 3; ++g)
        zero_f32<<<grid1(N_NODES * HID), TPB, 0, stream>>>(h[g], N_NODES * HID);

    auto gemm = [&](const float* A, const unsigned short* Bp, float* C,
                    const float* bias, int K, int N) {
        dim3 g((N_NODES + GTM - 1) / GTM, N / GTN);
        gemm_bf16_wmma<<<g, 256, 0, stream>>>(A, Bp, C, bias, N_NODES, K, N);
    };

    auto cell = [&](const float* xt, int xt_K) {
        // GCN1
        zero_f32<<<grid1(N_NODES * HID), TPB, 0, stream>>>(agg, N_NODES * HID);
        gemm(xt, pw_gcn1, xw, nullptr, xt_K, HID);
        edge_agg_kernel<<<N_EDGES, HID, 0, stream>>>(xw, src, dst, norm, agg);
        gcn_finish_kernel<<<grid1(N_NODES * HID), TPB, 0, stream>>>(agg, xw, deg_inv, gcn1_b, x1);
        // GCN2
        zero_f32<<<grid1(N_NODES * HID), TPB, 0, stream>>>(agg, N_NODES * HID);
        gemm(x1, pw_gcn2, xw, nullptr, HID, HID);
        edge_agg_kernel<<<N_EDGES, HID, 0, stream>>>(xw, src, dst, norm, agg);
        gcn_finish_kernel<<<grid1(N_NODES * HID), TPB, 0, stream>>>(agg, xw, deg_inv, gcn2_b, x2);
        // GRU stack
        const float* gin = x2;
        for (int g = 0; g < 3; ++g) {
            gemm(gin,  pw_wih[g], gi, bih[g], HID, 3 * HID);
            gemm(h[g], pw_whh[g], gh, bhh[g], HID, 3 * HID);
            gru_gate_kernel<<<grid1(N_NODES * HID), TPB, 0, stream>>>(gi, gh, h[g], h[g]);
            gin = h[g];
        }
    };

    // ---- encoder: 12 steps ----
    for (int t = 0; t < HIST; ++t)
        cell(x_seq + (size_t)t * N_NODES * IN_C, IN_C);

    // ---- decoder: 6 autoregressive steps ----
    const float* xd = x_seq + (size_t)(HIST - 1) * N_NODES * IN_C;
    for (int t = 0; t < HORIZON; ++t) {
        cell(xd, IN_C);
        float* y = out + (size_t)t * N_NODES * OUT_C;
        gemm(h[2], pw_proj, y, proj_b, HID, OUT_C);
        xd = y;   // next decoder input is this step's prediction
    }
}
